// PrimaryCapsule_79748952752184
// MI455X (gfx1250) — matrix operand order repaired and checked by reading, not compiled, not run on hardware
//
#include <hip/hip_runtime.h>
#include <hip/hip_bf16.h>
#include <stdint.h>

// ---------------------------------------------------------------------------
// PrimaryCapsule fused:  out(B,L,C) = squash( BN( W @ x ) )   [channel-last]
// GEMM via V_WMMA_F32_16X16X32_BF16 with bf16 hi/lo split (fp32-ish accuracy).
// W is BN-folded + bf16-split ONCE into d_ws by a prep kernel, then DMA'd
// into LDS with global_load_async_to_lds_b128 (ASYNCcnt path).
// ---------------------------------------------------------------------------

typedef float          v8f   __attribute__((ext_vector_type(8)));
typedef float          v4f   __attribute__((ext_vector_type(4)));
typedef __bf16         v16bf __attribute__((ext_vector_type(16)));
typedef unsigned short u16;
typedef u16            u16x8 __attribute__((ext_vector_type(8)));

#define B_DIM 32
#define C_DIM 256
#define L_DIM 8192
#define NT    32          // l-columns per workgroup
#define KP    264         // padded row pitch (halves): 256 + 8 to de-pattern banks

#define WBYTES   (2u * C_DIM * KP * 2u)            // Whi|Wlo in bf16, padded: 270336 B
#define WS_NEED  (WBYTES + C_DIM * 4u)             // + bias2[256]
#define CPY_IT   (WBYTES / (256u * 16u))           // 66 b128-copies per thread

union ABFrag { v16bf v; u16x8 h[2]; };
union Out8   { v8f   v; float f[8]; };

__device__ __forceinline__ u16 f2bf_rne(float f) {
    uint32_t u = __float_as_uint(f);
    u += 0x7FFFu + ((u >> 16) & 1u);          // round-to-nearest-even
    return (u16)(u >> 16);
}
__device__ __forceinline__ float bf2f(u16 h) {
    return __uint_as_float(((uint32_t)h) << 16);
}

// ---------------------------------------------------------------------------
// Prep kernel: A = diag(gamma/sqrt(var+eps)) * W  ->  bf16 hi/lo (padded KP
// layout, [Whi | Wlo] contiguous) in d_ws, plus folded bias2.
// ---------------------------------------------------------------------------
__global__ void __launch_bounds__(256)
caps_prep_kernel(const float* __restrict__ W,      const float* __restrict__ bias,
                 const float* __restrict__ gamma,  const float* __restrict__ beta,
                 const float* __restrict__ bn_mean,const float* __restrict__ bn_var,
                 u16* __restrict__ wsW, float* __restrict__ wsBias)
{
    const int i = blockIdx.x;        // output channel (row of W)
    const int c = threadIdx.x;       // input channel  (col of W)
    float inv = gamma[i] * rsqrtf(bn_var[i] + 1e-5f);
    float w   = W[i * C_DIM + c] * inv;
    u16   hi  = f2bf_rne(w);
    wsW[i * KP + c]               = hi;
    wsW[C_DIM * KP + i * KP + c]  = f2bf_rne(w - bf2f(hi));
    if (i == 0) {
        float invc = gamma[c] * rsqrtf(bn_var[c] + 1e-5f);
        wsBias[c]  = bias[c] * invc + beta[c] - bn_mean[c] * invc;
    }
}

// ---------------------------------------------------------------------------
// Main fused kernel.  PREP=true: async-DMA prebuilt W from d_ws into LDS.
//                     PREP=false: self-contained (convert W in-kernel).
// ---------------------------------------------------------------------------
template <bool PREP>
__global__ void __launch_bounds__(256, 1)
caps_fused_kernel(const float* __restrict__ x,      const float* __restrict__ W,
                  const float* __restrict__ bias,   const float* __restrict__ gamma,
                  const float* __restrict__ beta,   const float* __restrict__ bn_mean,
                  const float* __restrict__ bn_var, const u16* __restrict__ wsW,
                  const float* __restrict__ wsBias, float* __restrict__ out)
{
    extern __shared__ char smem[];
    u16*   Whi   = (u16*)smem;                 // C_DIM * KP  (BN-folded W, bf16 hi)
    u16*   Wlo   = Whi + C_DIM * KP;           // C_DIM * KP  (bf16 lo residual)
    u16*   Xhi   = Wlo + C_DIM * KP;           // NT * KP     (x tile, [n][k])
    u16*   Xlo   = Xhi + NT * KP;
    float* bias2 = (float*)(Xlo + NT * KP);    // C_DIM
    float* sumsq = bias2 + C_DIM;              // NT
    float* invs  = sumsq + NT;                 // C_DIM (fallback path only)

    const int tid   = threadIdx.x;
    const int lane  = tid & 31;
    const int wav   = tid >> 5;
    const int bIdx  = blockIdx.y;
    const int l0    = blockIdx.x * NT;

    if (PREP) {
        // -- W fill: async DMA, overlapped with the x-tile conversion below.
        {
            const unsigned ldsW = (unsigned)(uintptr_t)Whi + (unsigned)tid * 16u;
            unsigned long long src =
                (unsigned long long)(uintptr_t)wsW + (unsigned long long)tid * 16u;
            #pragma unroll 1
            for (unsigned it = 0; it < CPY_IT; ++it) {
                unsigned       d = ldsW + it * 4096u;
                unsigned long long s = src + (unsigned long long)it * 4096u;
                asm volatile("global_load_async_to_lds_b128 %0, %1, off"
                             :: "v"(d), "v"(s) : "memory");
            }
        }
        bias2[tid] = wsBias[tid];
        if (tid < NT) sumsq[tid] = 0.0f;
    } else {
        // ---- BN constants + in-kernel W fold/split (fallback) ----------
        {
            float inv = gamma[tid] * rsqrtf(bn_var[tid] + 1e-5f);
            invs[tid]  = inv;
            bias2[tid] = bias[tid] * inv + beta[tid] - bn_mean[tid] * inv;
            if (tid < NT) sumsq[tid] = 0.0f;
        }
        __syncthreads();
        for (int i = 0; i < C_DIM; ++i) {             // coalesced: row i, col tid
            float w  = W[i * C_DIM + tid] * invs[i];
            u16   hi = f2bf_rne(w);
            Whi[i * KP + tid] = hi;
            Wlo[i * KP + tid] = f2bf_rne(w - bf2f(hi));
        }
    }

    // ---- x tile: load transposed [n][k], split f32 -> bf16 hi/lo --------
    {
        const int n  = tid & (NT - 1);
        const int c0 = tid >> 5;                      // 0..7
        for (int j = 0; j < C_DIM / 8; ++j) {
            int   c  = j * 8 + c0;
            float v  = x[((size_t)bIdx * C_DIM + c) * L_DIM + l0 + n];
            u16   hi = f2bf_rne(v);
            Xhi[n * KP + c] = hi;
            Xlo[n * KP + c] = f2bf_rne(v - bf2f(hi));
        }
    }
    if (PREP) {
        asm volatile("s_wait_asynccnt 0x0" ::: "memory");  // W DMA complete
    }
    __syncthreads();

    // ---- WMMA K-loop (no barriers, full K resident) ---------------------
    const int m0    = wav * 32;                // this wave: rows m0..m0+31
    const int mlane = lane & 15;
    const int aoff  = (lane >> 4) << 3;        // A: upper lanes hold K+8..15 / +24..31
    const int boff  = (lane >> 4) << 3;        // B: upper lanes hold K+8..15 / +24..31

    v8f acc[2][2];
    #pragma unroll
    for (int i = 0; i < 2; ++i)
        #pragma unroll
        for (int j = 0; j < 2; ++j)
            acc[i][j] = (v8f){0.f,0.f,0.f,0.f,0.f,0.f,0.f,0.f};

    #pragma unroll 1
    for (int ks = 0; ks < C_DIM / 32; ++ks) {
        const int kb = ks * 32;
        ABFrag aHi[2], aLo[2], bHi[2], bLo[2];
        #pragma unroll
        for (int i = 0; i < 2; ++i) {
            const u16* ph = Whi + (m0 + i * 16 + mlane) * KP + kb + aoff;
            const u16* pl = Wlo + (m0 + i * 16 + mlane) * KP + kb + aoff;
            aHi[i].h[0] = *(const u16x8*)(ph);
            aHi[i].h[1] = *(const u16x8*)(ph + 16);
            aLo[i].h[0] = *(const u16x8*)(pl);
            aLo[i].h[1] = *(const u16x8*)(pl + 16);
        }
        #pragma unroll
        for (int j = 0; j < 2; ++j) {
            const u16* ph = Xhi + (j * 16 + mlane) * KP + kb + boff;
            const u16* pl = Xlo + (j * 16 + mlane) * KP + kb + boff;
            bHi[j].h[0] = *(const u16x8*)(ph);
            bHi[j].h[1] = *(const u16x8*)(ph + 16);
            bLo[j].h[0] = *(const u16x8*)(pl);
            bLo[j].h[1] = *(const u16x8*)(pl + 16);
        }
        #pragma unroll
        for (int i = 0; i < 2; ++i)
            #pragma unroll
            for (int j = 0; j < 2; ++j) {
                acc[i][j] = __builtin_amdgcn_wmma_f32_16x16x32_bf16(
                    false, aHi[i].v, false, bHi[j].v, (short)0, acc[i][j], false, false);
                acc[i][j] = __builtin_amdgcn_wmma_f32_16x16x32_bf16(
                    false, aHi[i].v, false, bLo[j].v, (short)0, acc[i][j], false, false);
                acc[i][j] = __builtin_amdgcn_wmma_f32_16x16x32_bf16(
                    false, aLo[i].v, false, bHi[j].v, (short)0, acc[i][j], false, false);
            }
    }

    // ---- bias, per-column |y|^2 reduction, squash, store ----------------
    Out8 y[2][2];
    #pragma unroll
    for (int i = 0; i < 2; ++i)
        #pragma unroll
        for (int j = 0; j < 2; ++j)
            y[i][j].v = acc[i][j];

    #pragma unroll
    for (int j = 0; j < 2; ++j) {
        float s = 0.0f;
        #pragma unroll
        for (int i = 0; i < 2; ++i) {
            const float* bp = bias2 + m0 + i * 16 + aoff;   // 8 consecutive channels
            #pragma unroll
            for (int r = 0; r < 8; ++r) {
                float v = y[i][j].f[r] + bp[r];
                y[i][j].f[r] = v;
                s += v * v;
            }
        }
        atomicAdd(&sumsq[j * 16 + mlane], s);   // ds_add_f32 cross-wave reduce
    }
    __syncthreads();

    #pragma unroll
    for (int j = 0; j < 2; ++j) {
        float  sc    = sumsq[j * 16 + mlane];                       // |y|^2
        float  scale = sc / ((1.0f + sc) * (sqrtf(sc) + 1e-8f));    // squash
        size_t col   = (size_t)bIdx * L_DIM + l0 + j * 16 + mlane;  // (b,l)
        #pragma unroll
        for (int i = 0; i < 2; ++i) {
            float* po = out + col * C_DIM + m0 + i * 16 + aoff;
            v4f q0, q1;
            #pragma unroll
            for (int r = 0; r < 4; ++r) {
                q0[r] = y[i][j].f[r]     * scale;
                q1[r] = y[i][j].f[r + 4] * scale;
            }
            *(v4f*)(po)     = q0;   // 2x global_store_b128
            *(v4f*)(po + 4) = q1;
        }
    }
}

extern "C" void kernel_launch(void* const* d_in, const int* in_sizes, int n_in,
                              void* d_out, int out_size, void* d_ws, size_t ws_size,
                              hipStream_t stream) {
    (void)in_sizes; (void)n_in; (void)out_size;
    const float* x     = (const float*)d_in[0];
    const float* W     = (const float*)d_in[1];
    const float* b     = (const float*)d_in[2];
    const float* gamma = (const float*)d_in[3];
    const float* beta  = (const float*)d_in[4];
    const float* mean  = (const float*)d_in[5];
    const float* var   = (const float*)d_in[6];

    const size_t smem = (size_t)(2 * C_DIM * KP + 2 * NT * KP) * sizeof(u16)
                      + (size_t)(C_DIM + NT + C_DIM) * sizeof(float);   // ~299 KB

    dim3 grid(L_DIM / NT, B_DIM);   // 256 x 32 workgroups

    if (ws_size >= WS_NEED) {
        u16*   wsW    = (u16*)d_ws;
        float* wsBias = (float*)((char*)d_ws + WBYTES);
        caps_prep_kernel<<<dim3(C_DIM), 256, 0, stream>>>(W, b, gamma, beta, mean, var,
                                                          wsW, wsBias);
        hipFuncSetAttribute(reinterpret_cast<const void*>(&caps_fused_kernel<true>),
                            hipFuncAttributeMaxDynamicSharedMemorySize, (int)smem);
        caps_fused_kernel<true><<<grid, 256, smem, stream>>>(
            x, W, b, gamma, beta, mean, var, wsW, wsBias, (float*)d_out);
    } else {
        hipFuncSetAttribute(reinterpret_cast<const void*>(&caps_fused_kernel<false>),
                            hipFuncAttributeMaxDynamicSharedMemorySize, (int)smem);
        caps_fused_kernel<false><<<grid, 256, smem, stream>>>(
            x, W, b, gamma, beta, mean, var, nullptr, nullptr, (float*)d_out);
    }
}
